// MultiHeadAttention_2224793059838
// MI455X (gfx1250) — matrix operand order repaired and checked by reading, not compiled, not run on hardware
//
#include <hip/hip_runtime.h>
#include <hip/hip_bf16.h>

// ---------------------------------------------------------------------------
// MHA on gfx1250: bf16 WMMA projections + flash attention + output GEMM.
// B=4, S=2048, DIM=1024, H=16, D=64.
// v4: ping-pong double-buffered GEMM loops (no register rotation copies),
//     32x64 wave tiles, V/K prefetch overlapping softmax in attention.
// ---------------------------------------------------------------------------

#define BB   4
#define SS   2048
#define DIMM 1024
#define HH   16
#define DD   64
#define MROWS (BB * SS)          // 8192

typedef __attribute__((ext_vector_type(16))) __bf16 v16bf;
typedef __attribute__((ext_vector_type(8)))  float  v8f;

// ---- bf16 bit helpers ------------------------------------------------------
__device__ __forceinline__ unsigned short f32_to_bf16_bits(float f) {
    unsigned int u = __builtin_bit_cast(unsigned int, f);
    u += 0x7FFFu + ((u >> 16) & 1u);           // round to nearest even
    return (unsigned short)(u >> 16);
}

struct uint4x2 { uint4 lo, hi; };
__device__ __forceinline__ v16bf as_v16bf(uint4 lo, uint4 hi) {
    uint4x2 u{lo, hi};
    return __builtin_bit_cast(v16bf, u);       // packed bf16 == v16bf bit layout
}

// ---- WMMA fragment loaders (packed bf16 source, zero ALU) ------------------
// A fragment (16x32, MxK), source row-major.
// lane<16: row=lane,    K = {0..7, 16..23}
// lane>=16: row=lane-16, K = {8..15, 24..31}
__device__ __forceinline__ v16bf load_a_bf16(const unsigned short* src, int ld, int lane) {
    const int row = lane & 15;
    const int k0  = (lane < 16) ? 0 : 8;
    const uint4* p = (const uint4*)(src + (size_t)row * ld + k0);
    return as_v16bf(p[0], p[2]);               // +0 and +16 elements
}
// B fragment (32x16, KxN), source row-major [K][N]: col = lane&15, K = {0..7,16..23} / {8..15,24..31}.
struct u16x16 { unsigned short e[16]; };
__device__ __forceinline__ v16bf load_b_bf16(const unsigned short* src, int ld, int lane) {
    const int n  = lane & 15;
    const int k0 = (lane >> 4) * 8;
    u16x16 t;
#pragma unroll
    for (int s = 0; s < 8; ++s) {
        t.e[s]     = src[(size_t)(k0 + s) * ld + n];
        t.e[8 + s] = src[(size_t)(16 + k0 + s) * ld + n];
    }
    return __builtin_bit_cast(v16bf, t);
}

__device__ __forceinline__ v8f wmma_bf16(v16bf a, v16bf b, v8f c) {
    return __builtin_amdgcn_wmma_f32_16x16x32_bf16(
        false, a, false, b, (short)0, c, false, false);
}

// ---- GEMM fragment set: A (32 rows -> 2 frags) x B (64 cols -> 4 frags) ----
struct Frags { v16bf a0, a1, b0, b1, b2, b3; };

__device__ __forceinline__ Frags load_frags(const unsigned short* Xa,
                                            const unsigned short* Xb,
                                            const unsigned short* Wb,
                                            int kk, int lane) {
    Frags f;
    f.a0 = load_a_bf16(Xa + kk, DIMM, lane);
    f.a1 = load_a_bf16(Xb + kk, DIMM, lane);
    const unsigned short* Wk = Wb + (size_t)kk * DIMM;
    f.b0 = load_b_bf16(Wk +  0, DIMM, lane);
    f.b1 = load_b_bf16(Wk + 16, DIMM, lane);
    f.b2 = load_b_bf16(Wk + 32, DIMM, lane);
    f.b3 = load_b_bf16(Wk + 48, DIMM, lane);
    return f;
}
__device__ __forceinline__ void consume_frags(const Frags& f, v8f acc0[4], v8f acc1[4]) {
    acc0[0] = wmma_bf16(f.a0, f.b0, acc0[0]);
    acc1[0] = wmma_bf16(f.a1, f.b0, acc1[0]);
    acc0[1] = wmma_bf16(f.a0, f.b1, acc0[1]);
    acc1[1] = wmma_bf16(f.a1, f.b1, acc1[1]);
    acc0[2] = wmma_bf16(f.a0, f.b2, acc0[2]);
    acc1[2] = wmma_bf16(f.a1, f.b2, acc1[2]);
    acc0[3] = wmma_bf16(f.a0, f.b3, acc0[3]);
    acc1[3] = wmma_bf16(f.a1, f.b3, acc1[3]);
}

// ---------------------------------------------------------------------------
// One-time f32 -> packed bf16 conversion (grid-stride, 8 elems/thread/iter).
// ---------------------------------------------------------------------------
__global__ __launch_bounds__(256)
void cvt_f32_bf16_kernel(const float* __restrict__ src,
                         unsigned short* __restrict__ dst, int n8) {
    int i = (int)(blockIdx.x * blockDim.x + threadIdx.x);
    const int stride = (int)(gridDim.x * blockDim.x);
    for (; i < n8; i += stride) {
        float4 a = ((const float4*)src)[2 * i];
        float4 b = ((const float4*)src)[2 * i + 1];
        uint4 o;
        o.x = (unsigned)f32_to_bf16_bits(a.x) | ((unsigned)f32_to_bf16_bits(a.y) << 16);
        o.y = (unsigned)f32_to_bf16_bits(a.z) | ((unsigned)f32_to_bf16_bits(a.w) << 16);
        o.z = (unsigned)f32_to_bf16_bits(b.x) | ((unsigned)f32_to_bf16_bits(b.y) << 16);
        o.w = (unsigned)f32_to_bf16_bits(b.z) | ((unsigned)f32_to_bf16_bits(b.w) << 16);
        ((uint4*)dst)[i] = o;
    }
}

// ---------------------------------------------------------------------------
// Projection: dst(bf16) = split_heads(X @ W + bias)
//   mode 0 -> dst[b][h][s][d]   (Q, V)
//   mode 1 -> dst[b][h][d][s]   (K transposed for QK^T B-fragments)
// One wave computes a 32x64 tile; ping-pong pipelined over K. 8 waves/block.
// ---------------------------------------------------------------------------
__global__ __launch_bounds__(256)
void proj_kernel(const unsigned short* __restrict__ X,
                 const unsigned short* __restrict__ W,
                 const float* __restrict__ bias,
                 unsigned short* __restrict__ dst, int mode) {
    const int gw   = (int)((blockIdx.x * blockDim.x + threadIdx.x) >> 5);
    const int lane = (int)(threadIdx.x & 31);
    const int tilesN = DIMM / 64;                 // 16
    const int m0 = (gw / tilesN) * 32;
    const int n0 = (gw % tilesN) * 64;

    const unsigned short* Xa = X + (size_t)m0 * DIMM;
    const unsigned short* Xb = X + (size_t)(m0 + 16) * DIMM;
    const unsigned short* Wb = W + n0;

    v8f acc0[4] = {v8f{}, v8f{}, v8f{}, v8f{}};
    v8f acc1[4] = {v8f{}, v8f{}, v8f{}, v8f{}};

    // ping-pong: two fragment sets, refilled in place after consumption
    Frags s0 = load_frags(Xa, Xb, Wb, 0,  lane);
    Frags s1 = load_frags(Xa, Xb, Wb, 32, lane);

#pragma unroll 1
    for (int kk = 64; kk < DIMM; kk += 64) {
        consume_frags(s0, acc0, acc1);
        s0 = load_frags(Xa, Xb, Wb, kk, lane);
        consume_frags(s1, acc0, acc1);
        s1 = load_frags(Xa, Xb, Wb, kk + 32, lane);
    }
    consume_frags(s0, acc0, acc1);
    consume_frags(s1, acc0, acc1);

    const int colLane = lane & 15;
    const int rowOff  = (lane >> 4) * 8;
#pragma unroll
    for (int half = 0; half < 2; ++half) {
        v8f* acc = half ? acc1 : acc0;
#pragma unroll
        for (int t = 0; t < 4; ++t) {
#pragma unroll
            for (int r = 0; r < 8; ++r) {
                const int m = m0 + half * 16 + r + rowOff;
                const int n = n0 + t * 16 + colLane;
                const float v = acc[t][r] + bias[n];
                const int bi = m >> 11, s = m & (SS - 1);
                const int h  = n >> 6,  d = n & (DD - 1);
                size_t off = (mode == 0)
                    ? ((((size_t)bi * HH + h) * SS + s) * DD + d)
                    : ((((size_t)bi * HH + h) * DD + d) * SS + s);
                dst[off] = f32_to_bf16_bits(v);
            }
        }
    }
}

// ---------------------------------------------------------------------------
// Flash attention. Block = 128 threads = 4 waves; each wave owns 16 q rows.
// Grid = B*H*(S/64). Streams KV in tiles of 32 with online softmax.
// V loads (current tile) and K loads (next tile) are issued right after the
// QK^T WMMAs so they overlap the softmax / LDS transpose.
// ---------------------------------------------------------------------------
__global__ __launch_bounds__(128)
void attn_kernel(const unsigned short* __restrict__ qh,
                 const unsigned short* __restrict__ kT,
                 const unsigned short* __restrict__ vh,
                 const float* __restrict__ mask,
                 unsigned short* __restrict__ ctx) {
    const int qtiles = SS / 64;                    // 32
    const int bh = (int)blockIdx.x / qtiles;       // 0..63
    const int qt = (int)blockIdx.x % qtiles;
    const int b  = bh / HH;
    const int h  = bh % HH;
    const int wave = (int)(threadIdx.x >> 5);
    const int lane = (int)(threadIdx.x & 31);
    const int q0   = qt * 64 + wave * 16;

    const unsigned short* qbase = qh + (((size_t)bh) * SS + q0) * DD;
    const unsigned short* kbase = kT + ((size_t)bh) * DD * SS;
    const unsigned short* vbase = vh + ((size_t)bh) * SS * DD;
    const float*          mrow  = mask + (size_t)b * SS;

    __shared__ __align__(16) unsigned short lds[4][16 * 32];  // 1KB per wave
    unsigned short* pbuf = lds[wave];

    // Q A-fragments (16x64 -> two K=32 fragments), loaded once
    v16bf aq0 = load_a_bf16(qbase,      DD, lane);
    v16bf aq1 = load_a_bf16(qbase + 32, DD, lane);

    v8f o[4] = {v8f{}, v8f{}, v8f{}, v8f{}};
    float mrun[8], lrun[8];
#pragma unroll
    for (int r = 0; r < 8; ++r) { mrun[r] = -INFINITY; lrun[r] = 0.0f; }

    const float scale = 0.125f;  // 1/sqrt(64)
    const int colLane = lane & 15;

    // prologue: K^T fragments for kv = 0
    v16bf kf0 = load_b_bf16(kbase + 0,                    SS, lane);
    v16bf kf1 = load_b_bf16(kbase + 16,                   SS, lane);
    v16bf kf2 = load_b_bf16(kbase + (size_t)32 * SS,      SS, lane);
    v16bf kf3 = load_b_bf16(kbase + (size_t)32 * SS + 16, SS, lane);

#pragma unroll 1
    for (int kv = 0; kv < SS; kv += 32) {
        // ---- S tile = Q (16x64) x K^T (64x32), two 16x16 halves ----
        v8f cs0 = v8f{}, cs1 = v8f{};
        cs0 = wmma_bf16(aq0, kf0, cs0);
        cs1 = wmma_bf16(aq0, kf1, cs1);
        cs0 = wmma_bf16(aq1, kf2, cs0);
        cs1 = wmma_bf16(aq1, kf3, cs1);

        // ---- prefetch: V fragments for this tile ----
        v16bf vf0 = load_b_bf16(vbase + (size_t)kv * DD +  0, DD, lane);
        v16bf vf1 = load_b_bf16(vbase + (size_t)kv * DD + 16, DD, lane);
        v16bf vf2 = load_b_bf16(vbase + (size_t)kv * DD + 32, DD, lane);
        v16bf vf3 = load_b_bf16(vbase + (size_t)kv * DD + 48, DD, lane);
        // ---- prefetch: K^T fragments for next tile (uniform branch, in place)
        if (kv + 32 < SS) {
            const unsigned short* kn = kbase + kv + 32;
            kf0 = load_b_bf16(kn,                        SS, lane);
            kf1 = load_b_bf16(kn + 16,                   SS, lane);
            kf2 = load_b_bf16(kn + (size_t)32 * SS,      SS, lane);
            kf3 = load_b_bf16(kn + (size_t)32 * SS + 16, SS, lane);
        }

        // additive mask (reference: logits + mask * -1e9)
        const float mk0 = mrow[kv + colLane]      * -1e9f;
        const float mk1 = mrow[kv + 16 + colLane] * -1e9f;

        // ---- online softmax per row (row lives across 16 lanes) ----
        v8f p0 = v8f{}, p1 = v8f{};
#pragma unroll
        for (int r = 0; r < 8; ++r) {
            float l0 = cs0[r] * scale + mk0;
            float l1 = cs1[r] * scale + mk1;
            float mx = fmaxf(l0, l1);
#pragma unroll
            for (int off = 8; off; off >>= 1)
                mx = fmaxf(mx, __shfl_xor(mx, off, 16));
            const float nm  = fmaxf(mrun[r], mx);
            const float e0  = __expf(l0 - nm);
            const float e1  = __expf(l1 - nm);
            float sum = e0 + e1;
#pragma unroll
            for (int off = 8; off; off >>= 1)
                sum += __shfl_xor(sum, off, 16);
            const float corr = __expf(mrun[r] - nm);
            lrun[r] = lrun[r] * corr + sum;
            mrun[r] = nm;
#pragma unroll
            for (int t = 0; t < 4; ++t) o[t][r] *= corr;
            p0[r] = e0;
            p1[r] = e1;
        }

        // ---- C-layout -> A-fragment transpose of P through LDS ----
        const int prow = (lane >> 4) * 8;
#pragma unroll
        for (int r = 0; r < 8; ++r) {
            pbuf[(r + prow) * 32 + colLane]      = f32_to_bf16_bits(p0[r]);
            pbuf[(r + prow) * 32 + 16 + colLane] = f32_to_bf16_bits(p1[r]);
        }
        asm volatile("s_wait_dscnt 0" ::: "memory");
        v16bf ap = load_a_bf16(pbuf, 32, lane);

        // ---- ctx (16x64) += P (16x32) x V (32x64) ----
        o[0] = wmma_bf16(ap, vf0, o[0]);
        o[1] = wmma_bf16(ap, vf1, o[1]);
        o[2] = wmma_bf16(ap, vf2, o[2]);
        o[3] = wmma_bf16(ap, vf3, o[3]);
    }

    // ---- epilogue: normalize, merge heads into [B][S][DIM] bf16 ----
    const int rowOff = (lane >> 4) * 8;
#pragma unroll
    for (int r = 0; r < 8; ++r) {
        const float inv = 1.0f / lrun[r];
        const int s = q0 + r + rowOff;
#pragma unroll
        for (int t = 0; t < 4; ++t) {
            const float v = o[t][r] * inv;
            const size_t off = (((size_t)b * SS + s) * DIMM) + h * DD + t * 16 + colLane;
            ctx[off] = f32_to_bf16_bits(v);
        }
    }
}

// ---------------------------------------------------------------------------
// Output GEMM: out(f32) = ctx(bf16) @ Wo(bf16) + bo  (32x64 tile, ping-pong)
// ---------------------------------------------------------------------------
__global__ __launch_bounds__(256)
void out_gemm_kernel(const unsigned short* __restrict__ ctx,
                     const unsigned short* __restrict__ Wo,
                     const float* __restrict__ bo,
                     float* __restrict__ out) {
    const int gw   = (int)((blockIdx.x * blockDim.x + threadIdx.x) >> 5);
    const int lane = (int)(threadIdx.x & 31);
    const int tilesN = DIMM / 64;
    const int m0 = (gw / tilesN) * 32;
    const int n0 = (gw % tilesN) * 64;

    const unsigned short* Xa = ctx + (size_t)m0 * DIMM;
    const unsigned short* Xb = ctx + (size_t)(m0 + 16) * DIMM;
    const unsigned short* Wb = Wo + n0;

    v8f acc0[4] = {v8f{}, v8f{}, v8f{}, v8f{}};
    v8f acc1[4] = {v8f{}, v8f{}, v8f{}, v8f{}};

    Frags s0 = load_frags(Xa, Xb, Wb, 0,  lane);
    Frags s1 = load_frags(Xa, Xb, Wb, 32, lane);

#pragma unroll 1
    for (int kk = 64; kk < DIMM; kk += 64) {
        consume_frags(s0, acc0, acc1);
        s0 = load_frags(Xa, Xb, Wb, kk, lane);
        consume_frags(s1, acc0, acc1);
        s1 = load_frags(Xa, Xb, Wb, kk + 32, lane);
    }
    consume_frags(s0, acc0, acc1);
    consume_frags(s1, acc0, acc1);

    const int colLane = lane & 15;
    const int rowOff  = (lane >> 4) * 8;
#pragma unroll
    for (int half = 0; half < 2; ++half) {
        v8f* acc = half ? acc1 : acc0;
#pragma unroll
        for (int t = 0; t < 4; ++t) {
#pragma unroll
            for (int r = 0; r < 8; ++r) {
                const int m = m0 + half * 16 + r + rowOff;
                const int n = n0 + t * 16 + colLane;
                out[(size_t)m * DIMM + n] = acc[t][r] + bo[n];
            }
        }
    }
}

// ---------------------------------------------------------------------------
extern "C" void kernel_launch(void* const* d_in, const int* in_sizes, int n_in,
                              void* d_out, int out_size, void* d_ws, size_t ws_size,
                              hipStream_t stream) {
    const float* Q    = (const float*)d_in[0];
    const float* K    = (const float*)d_in[1];
    const float* V    = (const float*)d_in[2];
    const float* mask = (const float*)d_in[3];
    const float* Wq   = (const float*)d_in[4];
    const float* bq   = (const float*)d_in[5];
    const float* Wk   = (const float*)d_in[6];
    const float* bk   = (const float*)d_in[7];
    const float* Wv   = (const float*)d_in[8];
    const float* bv   = (const float*)d_in[9];
    const float* Wo   = (const float*)d_in[10];
    const float* bo   = (const float*)d_in[11];

    const size_t nX = (size_t)MROWS * DIMM;         // 8M elems per activation
    const size_t nW = (size_t)DIMM * DIMM;          // 1M elems per weight

    unsigned short* ws  = (unsigned short*)d_ws;
    unsigned short* Qbf = ws;                       // reused as ctx after projections
    unsigned short* Kbf = Qbf + nX;
    unsigned short* Vbf = Kbf + nX;
    unsigned short* Wqb = Vbf + nX;
    unsigned short* Wkb = Wqb + nW;
    unsigned short* Wvb = Wkb + nW;
    unsigned short* Wob = Wvb + nW;
    unsigned short* qh  = Wob + nW;
    unsigned short* kTp = qh  + nX;
    unsigned short* vhp = kTp + nX;
    unsigned short* ctx = Qbf;                      // alias (Qbf dead after proj #1)

    // ---- one-time f32 -> bf16 conversions ----
    const int cvtBlocks = 1024;
    cvt_f32_bf16_kernel<<<cvtBlocks, 256, 0, stream>>>(Q,  Qbf, (int)(nX / 8));
    cvt_f32_bf16_kernel<<<cvtBlocks, 256, 0, stream>>>(K,  Kbf, (int)(nX / 8));
    cvt_f32_bf16_kernel<<<cvtBlocks, 256, 0, stream>>>(V,  Vbf, (int)(nX / 8));
    cvt_f32_bf16_kernel<<<cvtBlocks, 256, 0, stream>>>(Wq, Wqb, (int)(nW / 8));
    cvt_f32_bf16_kernel<<<cvtBlocks, 256, 0, stream>>>(Wk, Wkb, (int)(nW / 8));
    cvt_f32_bf16_kernel<<<cvtBlocks, 256, 0, stream>>>(Wv, Wvb, (int)(nW / 8));
    cvt_f32_bf16_kernel<<<cvtBlocks, 256, 0, stream>>>(Wo, Wob, (int)(nW / 8));

    // (MROWS/32)*(DIM/64) = 4096 wave-tiles, 8 waves/block -> 512 blocks
    const int projBlocks = (MROWS / 32) * (DIMM / 64) / 8;
    proj_kernel<<<projBlocks, 256, 0, stream>>>(Qbf, Wqb, bq, qh,  0);
    proj_kernel<<<projBlocks, 256, 0, stream>>>(Kbf, Wkb, bk, kTp, 1);
    proj_kernel<<<projBlocks, 256, 0, stream>>>(Vbf, Wvb, bv, vhp, 0);

    // B*H*(S/64) = 2048 blocks of 4 waves
    attn_kernel<<<BB * HH * (SS / 64), 128, 0, stream>>>(qh, kTp, vhp, mask, ctx);

    out_gemm_kernel<<<projBlocks, 256, 0, stream>>>(ctx, Wob, bo, (float*)d_out);
}
